// AdvancedNKATFinetuner_6588479832577
// MI455X (gfx1250) — compile-verified, matrix operand order checked by reading, not run on hardware
//
#include <hip/hip_runtime.h>

// ---------------------------------------------------------------------------
// MI455X (gfx1250) fused KAN-MLP forward.
// GEMMs on v_wmma_f32_16x16x32_bf16 (fp32 accum); A tiles staged to LDS with
// CDNA5 async DMA (global_load_async_to_lds_b128 / s_wait_asynccnt); W tiles
// converted fp32->bf16 through registers; LDS double-buffered so tile k+1
// fetch overlaps tile k WMMAs.  LN + KAN-spline + generator mix fused.
// ---------------------------------------------------------------------------

typedef __bf16 bf16;
typedef bf16  v4bf  __attribute__((ext_vector_type(4)));
typedef bf16  v8bf  __attribute__((ext_vector_type(8)));
typedef bf16  v16bf __attribute__((ext_vector_type(16)));
typedef float v8f   __attribute__((ext_vector_type(8)));

#define BATCH 1024
#define TILE  128
#define KT    32
#define LDSPAD 40   // row stride in bf16 elems: 80B = 16B-aligned, 5-bank stride

union FragU { v16bf v; v8bf h[2]; };

__device__ __forceinline__ float clip1(float x)  { return fminf(fmaxf(x, -1.f), 1.f); }
__device__ __forceinline__ float clip10(float x) { return fminf(fmaxf(x, -10.f), 10.f); }

// Exact replica of the reference's partial Cox-de Boor recursion
// (knots = [-1,-1,-1,-1,-0.6,-0.2,0.2,0.6,1,1,1,1], n = 8, k = 3).
__device__ __forceinline__ float kan_spline(float t, const float* cp) {
  const float kn[12] = {-1.f,-1.f,-1.f,-1.f,-0.6f,-0.2f,0.2f,0.6f,1.f,1.f,1.f,1.f};
  float c[8];
#pragma unroll
  for (int i = 0; i < 8; ++i)
    c[i] = (t >= kn[i] && t < kn[i + 1]) ? 1.f : 0.f;
#pragma unroll
  for (int deg = 1; deg <= 3; ++deg) {
#pragma unroll
    for (int i = 0; i + deg < 8; ++i) {
      float d1 = kn[i + deg] - kn[i];
      float d2 = kn[i + deg + 1] - kn[i + 1];
      float t1 = (d1 > 1e-10f) ? ((t - kn[i]) / d1) * c[i] : 0.f;
      float t2 = (d2 > 1e-10f) ? ((kn[i + deg + 1] - t) / d2) * c[i + 1] : 0.f;
      c[i] = t1 + t2;
    }
  }
  float s = 0.f;
#pragma unroll
  for (int i = 0; i < 8; ++i) s += c[i] * cp[i];
  return s;
}

// ------------------------- LayerNorm (no affine) ---------------------------
__global__ __launch_bounds__(256) void ln_bf16_kernel(const float* __restrict__ x,
                                                      bf16* __restrict__ out) {
  __shared__ float s1[256], s2[256];
  const int row = blockIdx.x, tid = threadIdx.x;
  const float* xr = x + (size_t)row * 2048;
  float v[8], sum = 0.f, sq = 0.f;
#pragma unroll
  for (int i = 0; i < 8; ++i) {
    float t = xr[tid + i * 256];
    v[i] = t; sum += t; sq += t * t;
  }
  s1[tid] = sum; s2[tid] = sq; __syncthreads();
  for (int off = 128; off > 0; off >>= 1) {
    if (tid < off) { s1[tid] += s1[tid + off]; s2[tid] += s2[tid + off]; }
    __syncthreads();
  }
  const float mean = s1[0] * (1.f / 2048.f);
  const float var  = s2[0] * (1.f / 2048.f) - mean * mean;
  const float rstd = rsqrtf(var + 1e-5f);
  bf16* o = out + (size_t)row * 2048;
#pragma unroll
  for (int i = 0; i < 8; ++i)
    o[tid + i * 256] = (bf16)((v[i] - mean) * rstd);
}

// Stage one 128x32 A tile (bf16, async DMA to LDS) and one 128x32 W tile
// (fp32 -> bf16 through registers) into the given LDS buffers.
__device__ __forceinline__ void stage_tile(
    const bf16* __restrict__ A, const float* __restrict__ W,
    int K, size_t k0, int m0, int n0, int tid,
    bf16 (*la)[LDSPAD], bf16 (*lw)[LDSPAD]) {
  // A tile: 512 x 16B chunks, 2 async b128 copies per thread (ASYNCcnt).
#pragma unroll
  for (int i = 0; i < 2; ++i) {
    int c = tid + i * 256;
    int row = c >> 2, kc = (c & 3) * 8;
    unsigned lds = (unsigned)(unsigned long long)&la[row][kc];
    const bf16* g = A + (size_t)(m0 + row) * K + k0 + kc;
    asm volatile("global_load_async_to_lds_b128 %0, %1, off"
                 :: "v"(lds), "v"(g) : "memory");
  }
  // W tile: batch 4 float4 loads into regs, then convert + ds_store.
  float4 wv[4];
#pragma unroll
  for (int i = 0; i < 4; ++i) {
    int c = tid + i * 256;
    int row = c >> 3, kc = (c & 7) * 4;
    wv[i] = *(const float4*)(W + (size_t)(n0 + row) * K + k0 + kc);
  }
#pragma unroll
  for (int i = 0; i < 4; ++i) {
    int c = tid + i * 256;
    int row = c >> 3, kc = (c & 7) * 4;
    v4bf bq;
    bq[0] = (bf16)wv[i].x; bq[1] = (bf16)wv[i].y;
    bq[2] = (bf16)wv[i].z; bq[3] = (bf16)wv[i].w;
    *(v4bf*)&lw[row][kc] = bq;
  }
  // Prefetch the K-tile after next into L2 (global_prefetch_b8).
  __builtin_prefetch(W + (size_t)(n0 + (tid >> 1)) * K + k0 + KT + (tid & 1) * 16, 0, 0);
}

// ----------------- GEMM (out = A @ W^T + b) + fused epilogue ---------------
// A: [M,K] bf16 row-major.  W: [N,K] fp32 row-major.
// mode 0: KAN + generator mix -> bf16 outB.  mode 1: bias only -> fp32 outF.
__global__ __launch_bounds__(256) void gemm_kan_kernel(
    const bf16* __restrict__ A, const float* __restrict__ W,
    const float* __restrict__ bias, const float* __restrict__ cp,
    const float* __restrict__ rw, bf16* __restrict__ outB,
    float* __restrict__ outF, int N, int K, int mode) {
  __shared__ bf16 lA[2][TILE][LDSPAD];
  __shared__ bf16 lW[2][TILE][LDSPAD];

  const int tid   = threadIdx.x;
  const int lane  = tid & 31;          // wave32
  const int wave  = tid >> 5;          // 8 waves per block
  const int kh    = lane >> 4;         // K-half select (ISA 16-bit A layout)
  const int l16   = lane & 15;
  const int waveM = wave & 3;          // 4 x 2 wave grid -> 128x128 tile
  const int waveN = wave >> 2;
  const int m0 = blockIdx.y * TILE;
  const int n0 = blockIdx.x * TILE;

  v8f acc[2][4] = {};

  const int nk = K / KT;
  stage_tile(A, W, K, 0, m0, n0, tid, lA[0], lW[0]);

  for (int kt = 0; kt < nk; ++kt) {
    const int cur = kt & 1;
    // Drain this wave's async A-tile DMA, then make the tile visible to all
    // waves; the same barrier also guarantees the previous compute finished
    // reading the buffer we are about to overwrite.
    asm volatile("s_wait_asynccnt 0x0" ::: "memory");
    __syncthreads();
    if (kt + 1 < nk)
      stage_tile(A, W, K, (size_t)(kt + 1) * KT, m0, n0, tid,
                 lA[cur ^ 1], lW[cur ^ 1]);

    // Fragment loads per ISA 16-bit A-layout: lane holds row (M or N),
    // lane-half selects K octet: chunk0 = K[kh*8..+8), chunk1 = K[16+kh*8..+8).
    FragU af[2], bf[4];
#pragma unroll
    for (int fm = 0; fm < 2; ++fm) {
      int r = waveM * 32 + fm * 16 + l16;
      af[fm].h[0] = *(const v8bf*)&lA[cur][r][kh * 8];
      af[fm].h[1] = *(const v8bf*)&lA[cur][r][16 + kh * 8];
    }
#pragma unroll
    for (int fn = 0; fn < 4; ++fn) {
      int r = waveN * 64 + fn * 16 + l16;
      bf[fn].h[0] = *(const v8bf*)&lW[cur][r][kh * 8];
      bf[fn].h[1] = *(const v8bf*)&lW[cur][r][16 + kh * 8];
    }
#pragma unroll
    for (int fm = 0; fm < 2; ++fm)
#pragma unroll
      for (int fn = 0; fn < 4; ++fn)
        acc[fm][fn] = __builtin_amdgcn_wmma_f32_16x16x32_bf16(
            false, af[fm].v, false, bf[fn].v, (short)0, acc[fm][fn],
            false, false);
  }

  // ------------------------------ Epilogue --------------------------------
  // C/D layout: VGPR r, lane -> row = r + 8*(lane>>4), col = lane&15.
#pragma unroll
  for (int fn = 0; fn < 4; ++fn) {
    const int n  = n0 + waveN * 64 + fn * 16 + l16;
    const float bn = bias[n];
    if (mode == 0) {
      float cpl[8];
#pragma unroll
      for (int i = 0; i < 8; ++i) cpl[i] = cp[n * 9 + i];
      const float rwn = rw[n];
      const int g = (n >> 1) & 3;                 // generator index of the pair
      const float sgn = (lane & 1) ? 1.f : -1.f;  // g==1: comp0 uses -partner
#pragma unroll
      for (int fm = 0; fm < 2; ++fm) {
#pragma unroll
        for (int r = 0; r < 8; ++r) {
          float lin  = acc[fm][fn][r] + bn;
          float t    = tanhf(lin);
          float kanv = clip10(kan_spline(t, cpl) * rwn);
          // pair partner lives in the adjacent lane (feature n^1)
          float p = __shfl_xor(kanv, 1, 32);
          float comm = (g == 0) ? clip1(p)
                     : (g == 1) ? clip1(sgn * p)
                                : 0.f;
          float o = kanv + 0.05f * comm;
          int row = m0 + waveM * 32 + fm * 16 + r + 8 * kh;
          outB[(size_t)row * N + n] = (bf16)o;
        }
      }
    } else {
#pragma unroll
      for (int fm = 0; fm < 2; ++fm)
#pragma unroll
        for (int r = 0; r < 8; ++r) {
          int row = m0 + waveM * 32 + fm * 16 + r + 8 * kh;
          outF[(size_t)row * N + n] = acc[fm][fn][r] + bn;
        }
    }
  }
}

// ---------------------------------------------------------------------------
extern "C" void kernel_launch(void* const* d_in, const int* in_sizes, int n_in,
                              void* d_out, int out_size, void* d_ws, size_t ws_size,
                              hipStream_t stream) {
  const float* x    = (const float*)d_in[0];
  const float* W1   = (const float*)d_in[1];
  const float* b1   = (const float*)d_in[2];
  const float* cp1  = (const float*)d_in[3];
  const float* rw1  = (const float*)d_in[4];
  const float* W2   = (const float*)d_in[5];
  const float* b2   = (const float*)d_in[6];
  const float* cp2  = (const float*)d_in[7];
  const float* rw2  = (const float*)d_in[8];
  const float* W3   = (const float*)d_in[9];
  const float* b3   = (const float*)d_in[10];
  const float* cp3  = (const float*)d_in[11];
  const float* rw3  = (const float*)d_in[12];
  const float* Wo   = (const float*)d_in[13];
  const float* bo   = (const float*)d_in[14];

  bf16* act0 = (bf16*)d_ws;                                          // 8 MB
  bf16* act1 = (bf16*)((char*)d_ws + (size_t)BATCH * 4096 * sizeof(bf16));

  ln_bf16_kernel<<<BATCH, 256, 0, stream>>>(x, act0);                // [1024,2048] bf16

  // layer 1: [1024,2048] @ [4096,2048]^T -> [1024,4096]
  gemm_kan_kernel<<<dim3(4096 / TILE, BATCH / TILE), 256, 0, stream>>>(
      act0, W1, b1, cp1, rw1, act1, nullptr, 4096, 2048, 0);
  // layer 2: [1024,4096] @ [4096,4096]^T -> [1024,4096]
  gemm_kan_kernel<<<dim3(4096 / TILE, BATCH / TILE), 256, 0, stream>>>(
      act1, W2, b2, cp2, rw2, act0, nullptr, 4096, 4096, 0);
  // layer 3: [1024,4096] @ [2048,4096]^T -> [1024,2048]
  gemm_kan_kernel<<<dim3(2048 / TILE, BATCH / TILE), 256, 0, stream>>>(
      act0, W3, b3, cp3, rw3, act1, nullptr, 2048, 4096, 0);
  // output: [1024,2048] @ [2048,2048]^T + b -> fp32 d_out
  gemm_kan_kernel<<<dim3(2048 / TILE, BATCH / TILE), 256, 0, stream>>>(
      act1, Wo, bo, nullptr, nullptr, nullptr, (float*)d_out, 2048, 2048, 1);

  (void)in_sizes; (void)n_in; (void)out_size; (void)ws_size;
}
